// MultiHeadAttentionBlock_15436112462644
// MI455X (gfx1250) — compile-verified, matrix operand order checked by reading, not run on hardware
//
#include <hip/hip_runtime.h>
#include <hip/hip_bf16.h>

typedef __attribute__((ext_vector_type(16))) __bf16 v16bf;
typedef __attribute__((ext_vector_type(8)))  float  v8f;
typedef __attribute__((ext_vector_type(4)))  unsigned int u32x4;
typedef __attribute__((ext_vector_type(8)))  unsigned int u32x8;

#define WMMA_BF16(a, b, c) \
  __builtin_amdgcn_wmma_f32_16x16x32_bf16(false, (a), false, (b), (short)0, (c), false, false)

static constexpr int kB = 4, kS = 2048, kD = 1024, kH = 16, kDK = 64;
static constexpr int kM = kB * kS;      // 8192 rows
static constexpr int kTiles = kS / 16;  // 128 mask tiles per axis

__device__ inline void tensor_wait0() {
#if __has_builtin(__builtin_amdgcn_s_wait_tensorcnt)
  __builtin_amdgcn_s_wait_tensorcnt(0);
#else
  asm volatile("s_wait_tensorcnt 0x0" ::: "memory");
#endif
}

// ---- TDM: 2-D tile global -> LDS. D# packed per CDNA5 ISA 8.3/8.4. ----
// data_size_code: 0=1B, 1=2B, 2=4B
__device__ inline void tdm_load_2d(unsigned lds_byte_off, const void* gptr,
                                   unsigned data_size_code,
                                   unsigned tensor_d0, unsigned tensor_d1,
                                   unsigned tile_d0, unsigned tile_d1,
                                   unsigned long long row_stride_elems) {
  const unsigned long long ga = (unsigned long long)gptr;
  u32x4 g0;
  g0[0] = 1u;                                                 // count=1, user desc
  g0[1] = lds_byte_off;                                       // lds_addr
  g0[2] = (unsigned)(ga & 0xffffffffu);                       // global_addr[31:0]
  g0[3] = (unsigned)((ga >> 32) & 0x01ffffffu) | (2u << 30);  // addr[56:32] | type=2
  u32x8 g1;
  g1[0] = (data_size_code << 16);
  g1[1] = (tensor_d0 & 0xffffu) << 16;                        // tensor_dim0[15:0]
  g1[2] = (tensor_d0 >> 16) | ((tensor_d1 & 0xffffu) << 16);
  g1[3] = (tensor_d1 >> 16) | (tile_d0 << 16);                // tile_dim0
  g1[4] = tile_d1;                                            // tile_dim1 (dim2=0)
  g1[5] = (unsigned)(row_stride_elems & 0xffffffffu);         // dim0_stride lo
  g1[6] = (unsigned)((row_stride_elems >> 32) & 0xffffu);
  g1[7] = 0u;
  asm volatile("tensor_load_to_lds %0, %1" :: "s"(g0), "s"(g1) : "memory");
}

// ---- fragment loader (ISA 7.12.2 16-bit A layout; also B as column-per-lane) ----
template <typename T>
__device__ inline v16bf load_row_frag(const T* __restrict__ p, int half) {
  v16bf f;
#pragma unroll
  for (int v = 0; v < 8; ++v) {
    const int kb = ((v & 4) << 2) + ((v & 3) << 1) + (half << 3);
    f[2 * v]     = (__bf16)(float)p[kb];
    f[2 * v + 1] = (__bf16)(float)p[kb + 1];
  }
  return f;
}

// xor-shuffle within 32-lane group via ds_swizzle (and=0x1f, or=0, xor=XOR)
template <int XOR>
__device__ inline float swz_max(float x) {
  const float y = __int_as_float(
      __builtin_amdgcn_ds_swizzle(__float_as_int(x), 0x1f | (XOR << 10)));
  return fmaxf(x, y);
}

// ---- elementwise f32 -> bf16 (weights pre-conversion) ----
__global__ __launch_bounds__(256) void cvt_f32_to_bf16(const float* __restrict__ in,
                                                       __bf16* __restrict__ out, int n) {
  const int i = (blockIdx.x * 256 + threadIdx.x) * 4;
  if (i + 3 < n) {
    const float4 f = *(const float4*)(in + i);
    out[i]     = (__bf16)f.x;
    out[i + 1] = (__bf16)f.y;
    out[i + 2] = (__bf16)f.z;
    out[i + 3] = (__bf16)f.w;
  }
}

// ---- Out = X @ W^T. Block = 8 waves sharing one 64-col bf16 W tile staged
//      via TDM (double-buffered LDS); each wave owns a 32-row M slice. ----
template <typename TX, typename TO, bool TRANS_OUT>
__global__ __launch_bounds__(256) void gemm_xWT(const TX* __restrict__ X,
                                                const __bf16* __restrict__ W,
                                                TO* __restrict__ Out,
                                                int M, int N, int K) {
  __shared__ __bf16 ldsW[2][64 * 32];  // 2 x 4 KB bf16 W tiles (64 rows x 32 k)
  const int lane = threadIdx.x & 31;
  const int half = lane >> 4;
  const int mn   = lane & 15;
  const int w    = threadIdx.x >> 5;
  const int nt   = blockIdx.x & 15;    // N/64 = 16
  const int mblk = blockIdx.x >> 4;
  const int m0 = (mblk * 8 + w) << 5;
  const int n0 = nt << 6;

  v8f acc[2][4];
#pragma unroll
  for (int i = 0; i < 2; ++i)
#pragma unroll
    for (int j = 0; j < 4; ++j) acc[i][j] = v8f{};

  const TX* x0 = X + (size_t)(m0 + mn) * K;
  const TX* x1 = X + (size_t)(m0 + 16 + mn) * K;

  const int iters = K >> 5;  // 32
  if (w == 0)
    tdm_load_2d(0, W + (size_t)n0 * K, 1, K, N - n0, 32, 64, K);

  for (int it = 0; it < iters; ++it) {
    const int k0 = it << 5;
    if (w == 0) tensor_wait0();          // current buffer ready
    __syncthreads();                     // visible to all; prior reads done
    if (w == 0 && it + 1 < iters)
      tdm_load_2d(((it + 1) & 1) * (64 * 32 * 2),
                  W + (size_t)n0 * K + (k0 + 32), 1,
                  K - (k0 + 32), N - n0, 32, 64, K);
    if (it + 2 < iters) {
      __builtin_prefetch(x0 + k0 + 64, 0, 0);
      __builtin_prefetch(x1 + k0 + 64, 0, 0);
    }
    const __bf16* wt = &ldsW[it & 1][0];
    v16bf a0 = load_row_frag(x0 + k0, half);
    v16bf a1 = load_row_frag(x1 + k0, half);
    v16bf b0 = load_row_frag(wt + (size_t)mn * 32, half);
    v16bf b1 = load_row_frag(wt + (size_t)(16 + mn) * 32, half);
    v16bf b2 = load_row_frag(wt + (size_t)(32 + mn) * 32, half);
    v16bf b3 = load_row_frag(wt + (size_t)(48 + mn) * 32, half);
    acc[0][0] = WMMA_BF16(a0, b0, acc[0][0]);
    acc[0][1] = WMMA_BF16(a0, b1, acc[0][1]);
    acc[0][2] = WMMA_BF16(a0, b2, acc[0][2]);
    acc[0][3] = WMMA_BF16(a0, b3, acc[0][3]);
    acc[1][0] = WMMA_BF16(a1, b0, acc[1][0]);
    acc[1][1] = WMMA_BF16(a1, b1, acc[1][1]);
    acc[1][2] = WMMA_BF16(a1, b2, acc[1][2]);
    acc[1][3] = WMMA_BF16(a1, b3, acc[1][3]);
  }

#pragma unroll
  for (int i = 0; i < 2; ++i)
#pragma unroll
    for (int j = 0; j < 4; ++j)
#pragma unroll
      for (int e = 0; e < 8; ++e) {
        const int row = m0 + (i << 4) + e + (half << 3);
        const int col = n0 + (j << 4) + mn;
        if (TRANS_OUT)
          Out[(size_t)col * M + row] = (TO)acc[i][j][e];  // [N][M] layout
        else
          Out[(size_t)row * N + col] = (TO)acc[i][j][e];  // [M][N] layout
      }
}

// ---- per-16x16-tile mask summary: flag=1 iff tile is all nonzero ----
__global__ __launch_bounds__(256) void mask_tile_flags(const int* __restrict__ mask,
                                                       int* __restrict__ flags) {
  const int lane = threadIdx.x & 31;
  const int w = threadIdx.x >> 5;
  const int t = blockIdx.x * 8 + w;
  const int qt = t / kTiles, kt = t % kTiles;
  bool zero = false;
#pragma unroll
  for (int i = 0; i < 8; ++i) {
    const int e = lane * 8 + i;
    const int r = e >> 4, c = e & 15;
    zero |= (mask[(size_t)(qt * 16 + r) * kS + kt * 16 + c] == 0);
  }
  const unsigned long long any0 = __ballot(zero);
  if (lane == 0) flags[t] = (any0 == 0ull) ? 1 : 0;
}

// ---- flash attention: block = 8 waves, all on the same (b,h); each wave owns
//      16 query rows. K/V 32-key tiles staged via TDM in double-buffered LDS
//      shared by the whole block. P row-sums via a ones-matrix WMMA. ----
__global__ __launch_bounds__(256) void flash_attn(const __bf16* __restrict__ Qp,
                                                  const __bf16* __restrict__ Kp,
                                                  const __bf16* __restrict__ Vt,
                                                  const int* __restrict__ mask,
                                                  const int* __restrict__ flags,
                                                  __bf16* __restrict__ Ctx) {
  // LDS: K dbuf [2][32 keys][64 dk] | V dbuf [2][64 dk][32 keys] | pbuf [8][16][32]
  __shared__ __bf16 smem[2 * 2048 + 2 * 2048 + 8 * 512];  // 24 KB
  __bf16* const ldsK = smem;                  // +0      (byte off 0)
  __bf16* const ldsV = smem + 4096;           // +4096   (byte off 8192)
  const int lane = threadIdx.x & 31;
  const int half = lane >> 4;
  const int mn   = lane & 15;
  const int w    = threadIdx.x >> 5;
  __bf16* const pb = smem + 8192 + w * 512;   // per-wave 16x32 P staging
  const int wid  = blockIdx.x * 8 + w;
  const int q0 = (wid & 127) << 4;
  const int h  = (wid >> 7) & 15;             // uniform across block
  const int b  = wid >> 11;                   // uniform across block
  const int* frow = flags + (size_t)(q0 >> 4) * kTiles;

  const __bf16* qbase = Qp + ((size_t)(b * kS + q0 + mn)) * kD + h * kDK;
  const v16bf aq0 = load_row_frag(qbase, half);
  const v16bf aq1 = load_row_frag(qbase + 32, half);

  v16bf ones;
#pragma unroll
  for (int i = 0; i < 16; ++i) ones[i] = (__bf16)1.0f;

  v8f o0 = v8f{}, o1 = v8f{}, o2 = v8f{}, o3 = v8f{};
  float mrow[8], lrow[8];
#pragma unroll
  for (int e = 0; e < 8; ++e) { mrow[e] = -1e30f; lrow[e] = 0.f; }

  const __bf16* kg = Kp + ((size_t)b * kS) * kD + h * kDK;           // K rows
  const __bf16* vg = Vt + ((size_t)(h * kDK)) * kM + (size_t)b * kS; // Vt rows

  if (w == 0) {
    tdm_load_2d(0,    kg, 1, 64, 32, 64, 32, kD);   // K tile keys 0..31
    tdm_load_2d(8192, vg, 1, 32, 64, 32, 64, kM);   // V tile keys 0..31
  }

  for (int c0 = 0, it = 0; c0 < kS; c0 += 32, ++it) {
    const int buf = it & 1;
    if (w == 0) tensor_wait0();        // this iter's K/V tiles landed
    __syncthreads();                   // visible; prior-iter LDS reads done
    if (w == 0 && c0 + 32 < kS) {
      tdm_load_2d((buf ^ 1) * 4096,        kg + (size_t)(c0 + 32) * kD,
                  1, 64, 32, 64, 32, kD);
      tdm_load_2d(8192 + (buf ^ 1) * 4096, vg + (c0 + 32),
                  1, 32, 64, 32, 64, kM);
    }
    const __bf16* kt = ldsK + buf * 2048;
    const __bf16* vt = ldsV + buf * 2048;

    const int fcomb = frow[c0 >> 4] & frow[(c0 >> 4) + 1];
    const bool allones = __builtin_amdgcn_readfirstlane(fcomb) != 0;

    // ---- scores: two 16x16 tiles (keys c0..c0+31), K from LDS ----
    v8f s0 = v8f{}, s1 = v8f{};
#pragma unroll
    for (int jt = 0; jt < 2; ++jt) {
      const __bf16* kr = kt + (size_t)(jt * 16 + mn) * 64;
      v16bf bk0 = load_row_frag(kr, half);
      v16bf bk1 = load_row_frag(kr + 32, half);
      v8f c = v8f{};
      c = WMMA_BF16(aq0, bk0, c);
      c = WMMA_BF16(aq1, bk1, c);
#pragma unroll
      for (int e = 0; e < 8; ++e) c[e] *= 0.125f;        // 1/sqrt(DK=64)
      if (!allones) {                                     // rare path: exact mask
#pragma unroll
        for (int e = 0; e < 8; ++e) {
          const int qrow = q0 + e + (half << 3);
          const int key  = c0 + jt * 16 + mn;
          if (mask[(size_t)qrow * kS + key] == 0) c[e] = -1e9f;
        }
      }
      if (jt == 0) s0 = c; else s1 = c;
    }

    // ---- row max (ds_swizzle xor-reduction in 16-lane groups) ----
    float alpha[8];
#pragma unroll
    for (int e = 0; e < 8; ++e) {
      float r = fmaxf(s0[e], s1[e]);
      r = swz_max<1>(r); r = swz_max<2>(r); r = swz_max<4>(r); r = swz_max<8>(r);
      const float mnew = fmaxf(mrow[e], r);
      alpha[e] = __expf(mrow[e] - mnew);
      s0[e] = __expf(s0[e] - mnew);      // s now holds P
      s1[e] = __expf(s1[e] - mnew);
      mrow[e] = mnew;
    }

    // ---- stage P through LDS, reload as 16x32 A fragment ----
#pragma unroll
    for (int e = 0; e < 8; ++e) {
      pb[(size_t)(e + (half << 3)) * 32 + mn]      = (__bf16)s0[e];
      pb[(size_t)(e + (half << 3)) * 32 + 16 + mn] = (__bf16)s1[e];
    }
    __syncthreads();
    const v16bf ap = load_row_frag(pb + (size_t)mn * 32, half);

    // ---- row sums of P via ones-matrix WMMA (replicated per 16-lane group) ----
    const v8f rs = WMMA_BF16(ap, ones, v8f{});
#pragma unroll
    for (int e = 0; e < 8; ++e) {
      lrow[e] = lrow[e] * alpha[e] + rs[e];
      o0[e] *= alpha[e]; o1[e] *= alpha[e]; o2[e] *= alpha[e]; o3[e] *= alpha[e];
    }

    // ---- P (16x32) x V (32x64): V columns are contiguous LDS rows ----
    v16bf bv0 = load_row_frag(vt + (size_t)(mn)      * 32, half);
    v16bf bv1 = load_row_frag(vt + (size_t)(16 + mn) * 32, half);
    v16bf bv2 = load_row_frag(vt + (size_t)(32 + mn) * 32, half);
    v16bf bv3 = load_row_frag(vt + (size_t)(48 + mn) * 32, half);
    o0 = WMMA_BF16(ap, bv0, o0);
    o1 = WMMA_BF16(ap, bv1, o1);
    o2 = WMMA_BF16(ap, bv2, o2);
    o3 = WMMA_BF16(ap, bv3, o3);
  }

  __bf16* cb = Ctx + ((size_t)(b * kS + q0)) * kD + h * kDK;
#pragma unroll
  for (int e = 0; e < 8; ++e) {
    const float inv = lrow[e] > 0.f ? 1.f / lrow[e] : 0.f;
    const size_t ro = (size_t)(e + (half << 3)) * kD;
    cb[ro + mn]      = (__bf16)(o0[e] * inv);
    cb[ro + 16 + mn] = (__bf16)(o1[e] * inv);
    cb[ro + 32 + mn] = (__bf16)(o2[e] * inv);
    cb[ro + 48 + mn] = (__bf16)(o3[e] * inv);
  }
}

extern "C" void kernel_launch(void* const* d_in, const int* in_sizes, int n_in,
                              void* d_out, int out_size, void* d_ws, size_t ws_size,
                              hipStream_t stream) {
  const float* q    = (const float*)d_in[0];
  const float* k    = (const float*)d_in[1];
  const float* v    = (const float*)d_in[2];
  const int*   mask = (const int*)  d_in[3];
  const float* Wq   = (const float*)d_in[4];
  const float* Wk   = (const float*)d_in[5];
  const float* Wv   = (const float*)d_in[6];
  const float* Wo   = (const float*)d_in[7];
  float* out = (float*)d_out;

  const size_t NSD = (size_t)kM * kD;       // 8,388,608 elements
  const size_t NW  = (size_t)kD * kD;       // 1,048,576 elements
  __bf16* Qp  = (__bf16*)d_ws;              // 16 MB each
  __bf16* Kp  = Qp + NSD;
  __bf16* Vt  = Kp + NSD;                   // transposed [D][B*S]
  __bf16* Ctx = Vt + NSD;
  __bf16* Wqb = Ctx + NSD;                  // bf16 weights, 2 MB each
  __bf16* Wkb = Wqb + NW;
  __bf16* Wvb = Wkb + NW;
  __bf16* Wob = Wvb + NW;
  int* flags  = (int*)(Wob + NW);           // 128*128 ints = 64 KB

  mask_tile_flags<<<(kTiles * kTiles) / 8, 256, 0, stream>>>(mask, flags);

  const int cvtBlocks = (int)(NW / (256 * 4));  // 1024
  cvt_f32_to_bf16<<<cvtBlocks, 256, 0, stream>>>(Wq, Wqb, (int)NW);
  cvt_f32_to_bf16<<<cvtBlocks, 256, 0, stream>>>(Wk, Wkb, (int)NW);
  cvt_f32_to_bf16<<<cvtBlocks, 256, 0, stream>>>(Wv, Wvb, (int)NW);
  cvt_f32_to_bf16<<<cvtBlocks, 256, 0, stream>>>(Wo, Wob, (int)NW);

  const int gemmBlocks = (kM / 256) * (kD / 64);  // 32 * 16 = 512
  gemm_xWT<float, __bf16, false><<<gemmBlocks, 256, 0, stream>>>(q, Wqb, Qp, kM, kD, kD);
  gemm_xWT<float, __bf16, false><<<gemmBlocks, 256, 0, stream>>>(k, Wkb, Kp, kM, kD, kD);
  gemm_xWT<float, __bf16, true ><<<gemmBlocks, 256, 0, stream>>>(v, Wvb, Vt, kM, kD, kD);

  const int attnBlocks = kB * kH * (kS / 16) / 8;  // 1024
  flash_attn<<<attnBlocks, 256, 0, stream>>>(Qp, Kp, Vt, mask, flags, Ctx);

  gemm_xWT<__bf16, float, false><<<gemmBlocks, 256, 0, stream>>>(Ctx, Wob, out, kM, kD, kD);
}
